// PointNetPlusPlus_1760936591723
// MI455X (gfx1250) — matrix operand order repaired and checked by reading, not compiled, not run on hardware
//
#include <hip/hip_runtime.h>

// ---------------------------------------------------------------------------
// PointNet++ forward for MI455X (gfx1250, wave32, WMMA).
// All linear layers -> WMMA GEMM (v_wmma_f32_16x16x32_f16).
// Each wave computes a 16x(16*NT) output strip: the A fragment is loaded once
// per k-step and reused across NT B fragments (NT=4 for all wide layers).
// A matrices staged in f16 with K padded to multiples of 32 (zero fill),
// weights pre-transposed to f16 [Nt][Kp] so the inner loop is pure aligned
// b128 vector loads + WMMA (no predication).
// ---------------------------------------------------------------------------

typedef __attribute__((ext_vector_type(16))) _Float16 v16h;
typedef __attribute__((ext_vector_type(8)))  _Float16 v8h;
typedef __attribute__((ext_vector_type(8)))  float    v8f;

// Problem constants (match reference).
#define BCLOUDS 4
#define NPTS    4096
#define BN      (BCLOUDS * NPTS)
#define M1PER   (NPTS / 2)          // 2048 centers, level 1
#define M2PER   (NPTS / 8)          // 512 centers, level 2
#define M1TOT   (BCLOUDS * M1PER)   // 8192
#define M2TOT   (BCLOUDS * M2PER)   // 2048
#define KGRP    32
#define KINT    3

// ---------------------------------------------------------------------------
// WMMA GEMM:  Out[M,N] = act( A[M,Kp](f16) @ Wt[Nt,Kp]^T(f16) + bias[N] )
// Preconditions: M % 64 == 0, Kp % 32 == 0, Nt % (16*NT) == 0, A zero-padded
// beyond true K, Wt zero-padded beyond true K/N.
// 4 waves per block along M; each wave owns NT adjacent 16x16 N-tiles.
// Fragment layouts as used here (contiguous k order, not the ISA table order):
//   A: lane l, m=l&15, khalf=l>>4, elem e: k = khalf*16 + e
//      -> two contiguous 8-half (16B) runs per k-step.
//   B: n = l&15, elem e: k = khalf*16 + e -> one contiguous 16-half (32B) run.
//   C/D: elem v -> row v + 8*khalf, col l&15.
// ---------------------------------------------------------------------------
template <int NT, bool RELU, bool OUT_F32>
__global__ __launch_bounds__(128)
void gemm_wmma(const _Float16* __restrict__ A, const _Float16* __restrict__ Wt,
               const float* __restrict__ bias, void* __restrict__ Out,
               int M, int Kp, int N, int Nt)
{
    const int lane   = threadIdx.x & 31;
    const int wave   = threadIdx.x >> 5;
    const int mtile  = blockIdx.x * 4 + wave;
    const int ntile0 = blockIdx.y * NT;
    const int mlo    = lane & 15;
    const int khalf  = lane >> 4;
    const int row    = mtile * 16 + mlo;   // always < M (M % 64 == 0)

    const _Float16* arow = A + (size_t)row * Kp + khalf * 16;
    const _Float16* brow[NT];
#pragma unroll
    for (int t = 0; t < NT; ++t)
        brow[t] = Wt + ((size_t)(ntile0 + t) * 16 + mlo) * Kp + khalf * 16;

    v8f acc[NT];
#pragma unroll
    for (int t = 0; t < NT; ++t) acc[t] = (v8f){};

    for (int k0 = 0; k0 < Kp; k0 += 32) {
        v8h a0 = *reinterpret_cast<const v8h*>(arow + k0);        // k offs 0..7
        v8h a1 = *reinterpret_cast<const v8h*>(arow + k0 + 8);    // k offs 8..15
        v16h a = __builtin_shufflevector(a0, a1, 0, 1, 2, 3, 4, 5, 6, 7,
                                                 8, 9, 10, 11, 12, 13, 14, 15);
#pragma unroll
        for (int t = 0; t < NT; ++t) {
            v16h b = *reinterpret_cast<const v16h*>(brow[t] + k0); // k offs 0..15
#if defined(__AMDGCN__)
            acc[t] = __builtin_amdgcn_wmma_f32_16x16x32_f16(
                /*neg_a=*/false, a, /*neg_b=*/false, b,
                /*c_mod=*/(short)0, acc[t], /*reuse_a=*/false, /*reuse_b=*/false);
#else
            (void)a; (void)b;
#endif
        }
    }

#pragma unroll
    for (int t = 0; t < NT; ++t) {
        const int col = (ntile0 + t) * 16 + mlo;  // < Nt; may exceed N only if OUT_F32
        const float bs = (col < N) ? bias[col] : 0.f;
#pragma unroll
        for (int v = 0; v < 8; ++v) {
            int r = mtile * 16 + v + 8 * khalf;
            float o = acc[t][v] + bs;
            if (RELU) o = fmaxf(o, 0.f);
            if (OUT_F32) {
                if (col < N) ((float*)Out)[(size_t)r * N + col] = o;
            } else {
                ((_Float16*)Out)[(size_t)r * N + col] = (_Float16)o; // N % 16 == 0
            }
        }
    }
}

// Weight prep: W[K,N] f32 -> Wt[Nt,Kp] f16 (transposed, zero-padded).
__global__ void prep_w(const float* __restrict__ W, int K, int N, int Kp, int Nt,
                       _Float16* __restrict__ Wt)
{
    int id = blockIdx.x * blockDim.x + threadIdx.x;
    if (id >= Nt * Kp) return;
    int n = id / Kp, k = id % Kp;
    Wt[id] = (k < K && n < N) ? (_Float16)W[(size_t)k * N + n] : (_Float16)0.f;
}

// f0 = concat(x[.,4], pos[.,3]) as f16
__global__ void build_f0(const float* __restrict__ x, const float* __restrict__ pos,
                         _Float16* __restrict__ f0, int n)
{
    int i = blockIdx.x * blockDim.x + threadIdx.x;
    if (i >= n) return;
    f0[i * 7 + 0] = (_Float16)x[i * 4 + 0];
    f0[i * 7 + 1] = (_Float16)x[i * 4 + 1];
    f0[i * 7 + 2] = (_Float16)x[i * 4 + 2];
    f0[i * 7 + 3] = (_Float16)x[i * 4 + 3];
    f0[i * 7 + 4] = (_Float16)pos[i * 3 + 0];
    f0[i * 7 + 5] = (_Float16)pos[i * 3 + 1];
    f0[i * 7 + 6] = (_Float16)pos[i * 3 + 2];
}

// ---------------------------------------------------------------------------
// Farthest point sampling: one block per cloud, distance array in LDS.
// ---------------------------------------------------------------------------
__global__ void fps_kernel(const float* __restrict__ pos, int n, int m,
                           int* __restrict__ idx_out)
{
    __shared__ float sd[4096];
    __shared__ float rv[256];
    __shared__ int   ri[256];
    const int b = blockIdx.x;
    const float* p = pos + (size_t)b * n * 3;
    int* out = idx_out + (size_t)b * m;
    const int t = threadIdx.x;
    const int T = blockDim.x;

    const float x0 = p[0], y0 = p[1], z0 = p[2];
    for (int i = t; i < n; i += T) {
        float dx = p[i * 3 + 0] - x0, dy = p[i * 3 + 1] - y0, dz = p[i * 3 + 2] - z0;
        sd[i] = dx * dx + dy * dy + dz * dz;
    }
    if (t == 0) out[0] = 0;

    for (int s = 1; s < m; ++s) {
        __syncthreads();
        float bv = -1.f; int bidx = 0;
        for (int i = t; i < n; i += T) {
            float v = sd[i];
            if (v > bv) { bv = v; bidx = i; }
        }
        rv[t] = bv; ri[t] = bidx;
        __syncthreads();
        for (int o = T >> 1; o > 0; o >>= 1) {
            if (t < o) {
                float v2 = rv[t + o]; int i2 = ri[t + o];
                if (v2 > rv[t] || (v2 == rv[t] && i2 < ri[t])) { rv[t] = v2; ri[t] = i2; }
            }
            __syncthreads();
        }
        const int nxt = ri[0];
        if (t == 0) out[s] = nxt;
        const float nx = p[nxt * 3 + 0], ny = p[nxt * 3 + 1], nz = p[nxt * 3 + 2];
        for (int i = t; i < n; i += T) {
            float dx = p[i * 3 + 0] - nx, dy = p[i * 3 + 1] - ny, dz = p[i * 3 + 2] - nz;
            float d = dx * dx + dy * dy + dz * dz;
            sd[i] = fminf(sd[i], d);
        }
    }
}

// Gather f32 rows by per-cloud local indices.
__global__ void gather_rows_f32(const float* __restrict__ src, const int* __restrict__ idx,
                                int n_idx, int n_src, int F, float* __restrict__ out,
                                int total)
{
    int id = blockIdx.x * blockDim.x + threadIdx.x;
    if (id >= total) return;
    int f = id % F, g = id / F, b = g / n_idx;
    out[id] = src[((size_t)b * n_src + idx[g]) * F + f];
}

// ---------------------------------------------------------------------------
// Brute-force kNN, insertion top-k, tie-break by index (matches top_k).
// ---------------------------------------------------------------------------
template <int KK>
__global__ void knn_kernel(const float* __restrict__ src, const float* __restrict__ qry,
                           int n_src, int n_qry, int n_clouds,
                           int* __restrict__ oidx, float* __restrict__ od2)
{
    int q = blockIdx.x * blockDim.x + threadIdx.x;
    if (q >= n_clouds * n_qry) return;
    int b = q / n_qry;
    const float* s = src + (size_t)b * n_src * 3;
    const float qx = qry[q * 3 + 0], qy = qry[q * 3 + 1], qz = qry[q * 3 + 2];

    float bd[KK]; int bi[KK];
#pragma unroll
    for (int j = 0; j < KK; ++j) { bd[j] = 3.4e38f; bi[j] = 0; }

    for (int i = 0; i < n_src; ++i) {
        float dx = s[i * 3 + 0] - qx, dy = s[i * 3 + 1] - qy, dz = s[i * 3 + 2] - qz;
        float d = dx * dx + dy * dy + dz * dz;
        if (d < bd[KK - 1]) {
            int j = KK - 1;
            while (j > 0 && d < bd[j - 1]) { bd[j] = bd[j - 1]; bi[j] = bi[j - 1]; --j; }
            bd[j] = d; bi[j] = i;
        }
    }
    for (int j = 0; j < KK; ++j) { oidx[(size_t)q * KK + j] = bi[j]; od2[(size_t)q * KK + j] = bd[j]; }
}

// H[c*Kg+j, :] = concat(x_src[idx], pos_src[idx]-pos_c[c], zeros) with row
// stride ld (>= F+3), zero-padding the tail so GEMM needs no K bounds checks.
__global__ void sa_gather(const _Float16* __restrict__ xsrc, const float* __restrict__ psrc,
                          const float* __restrict__ pc, const int* __restrict__ kidx,
                          int n_src, int n_c, int n_clouds, int Kg, int F, int ld,
                          _Float16* __restrict__ H)
{
    int id = blockIdx.x * blockDim.x + threadIdx.x;
    int total = n_clouds * n_c * Kg;
    if (id >= total) return;
    int cg = id / Kg;
    int b = cg / n_c;
    int srcr = b * n_src + kidx[id];
    _Float16* h = H + (size_t)id * ld;
    const _Float16* xs = xsrc + (size_t)srcr * F;
    for (int f = 0; f < F; ++f) h[f] = xs[f];
    h[F + 0] = (_Float16)(psrc[srcr * 3 + 0] - pc[cg * 3 + 0]);
    h[F + 1] = (_Float16)(psrc[srcr * 3 + 1] - pc[cg * 3 + 1]);
    h[F + 2] = (_Float16)(psrc[srcr * 3 + 2] - pc[cg * 3 + 2]);
    for (int f = F + 3; f < ld; ++f) h[f] = (_Float16)0.f;
}

// out[c,f] = max_j in[c*Kg+j, f]
__global__ void maxagg(const _Float16* __restrict__ in, int Kg, int F, int nC,
                       _Float16* __restrict__ out)
{
    int id = blockIdx.x * blockDim.x + threadIdx.x;
    if (id >= nC * F) return;
    int c = id / F, f = id % F;
    float m = -3.4e38f;
    for (int j = 0; j < Kg; ++j)
        m = fmaxf(m, (float)in[((size_t)c * Kg + j) * F + f]);
    out[id] = (_Float16)m;
}

// out[r,:] = concat(a[r,:Fa], b2[r,:Fb], zeros) with row stride ld.
__global__ void concat2(const _Float16* __restrict__ a, int Fa,
                        const _Float16* __restrict__ b2, int Fb,
                        int n, int ld, _Float16* __restrict__ out)
{
    int id = blockIdx.x * blockDim.x + threadIdx.x;
    if (id >= n * ld) return;
    int r = id / ld, f = id % ld;
    _Float16 v = (_Float16)0.f;
    if (f < Fa)           v = a[(size_t)r * Fa + f];
    else if (f < Fa + Fb) v = b2[(size_t)r * Fb + (f - Fa)];
    out[id] = v;
}

// knn_interpolate with k=3, inverse-squared-distance weights.
__global__ void interp3(const _Float16* __restrict__ xsrc, const int* __restrict__ idx,
                        const float* __restrict__ d2, int n_src, int n_dst, int n_clouds,
                        int F, _Float16* __restrict__ out)
{
    int id = blockIdx.x * blockDim.x + threadIdx.x;
    if (id >= n_clouds * n_dst * F) return;
    int g = id / F, f = id % F, b = g / n_dst;
    float w0 = 1.f / fmaxf(d2[g * 3 + 0], 1e-16f);
    float w1 = 1.f / fmaxf(d2[g * 3 + 1], 1e-16f);
    float w2 = 1.f / fmaxf(d2[g * 3 + 2], 1e-16f);
    float ws = w0 + w1 + w2;
    size_t base = (size_t)b * n_src;
    float v = (w0 * (float)xsrc[(base + idx[g * 3 + 0]) * F + f] +
               w1 * (float)xsrc[(base + idx[g * 3 + 1]) * F + f] +
               w2 * (float)xsrc[(base + idx[g * 3 + 2]) * F + f]) / ws;
    out[id] = (_Float16)v;
}

// ---------------------------------------------------------------------------
// Orchestration
// ---------------------------------------------------------------------------
static inline int cdiv_i(int a, int b) { return (a + b - 1) / b; }

extern "C" void kernel_launch(void* const* d_in, const int* in_sizes, int n_in,
                              void* d_out, int out_size, void* d_ws, size_t ws_size,
                              hipStream_t stream)
{
    (void)in_sizes; (void)n_in; (void)out_size; (void)ws_size;

    const float* x   = (const float*)d_in[0];   // [BN,4]
    const float* pos = (const float*)d_in[1];   // [BN,3]
    // d_in[2] = batch (structure baked into constants)

    const float* Wf[14], *bf[14];
    for (int i = 0; i < 14; ++i) {
        Wf[i] = (const float*)d_in[3 + 2 * i];
        bf[i] = (const float*)d_in[4 + 2 * i];
    }
    // layer table: {K, N, Kp, Nt}
    // 0 sa1_l1 10->64 | 1 sa1_l2 64->64 | 2 sa1_g 64->128
    // 3 sa2_l1 131->128 | 4 sa2_l2 128->256 | 5 sa2_g 256->512
    // 6 fp2_1 640->256 | 7 fp2_2 256->256 | 8 fp1_1 263->128 | 9 fp1_2 128->128
    // 10 sem_1 128->64 | 11 sem_2 64->8 | 12 ins_1 128->64 | 13 ins_2 64->64
    const int LK[14]  = {10, 64, 64, 131, 128, 256, 640, 256, 263, 128, 128, 64, 128, 64};
    const int LN[14]  = {64, 64, 128, 128, 256, 512, 256, 256, 128, 128, 64, 8, 64, 64};
    const int LKP[14] = {32, 64, 64, 160, 128, 256, 640, 256, 288, 128, 128, 64, 128, 64};
    const int LNT[14] = {64, 64, 128, 128, 256, 512, 256, 256, 128, 128, 64, 16, 64, 64};

    // -------- workspace bump allocator (256B aligned) --------
    size_t off = 0;
    char* wsb = (char*)d_ws;
    auto alloc = [&](size_t bytes) -> void* {
        off = (off + 255) & ~(size_t)255;
        void* p = wsb + off;
        off += bytes;
        return p;
    };

    // transposed / padded f16 weights
    _Float16* Wt[14];
    for (int i = 0; i < 14; ++i)
        Wt[i] = (_Float16*)alloc((size_t)LNT[i] * LKP[i] * 2);

    _Float16* f0h  = (_Float16*)alloc((size_t)BN * 7 * 2);
    int*   idx1  = (int*)  alloc((size_t)M1TOT * 4);
    float* pos1  = (float*)alloc((size_t)M1TOT * 3 * 4);
    int*   kidx1 = (int*)  alloc((size_t)M1TOT * KGRP * 4);
    float* kd1   = (float*)alloc((size_t)M1TOT * KGRP * 4);
    int*   idx2  = (int*)  alloc((size_t)M2TOT * 4);
    float* pos2  = (float*)alloc((size_t)M2TOT * 3 * 4);
    int*   kidx2 = (int*)  alloc((size_t)M2TOT * KGRP * 4);
    float* kd2   = (float*)alloc((size_t)M2TOT * KGRP * 4);
    int*   ii2   = (int*)  alloc((size_t)M1TOT * KINT * 4);
    float* dd2   = (float*)alloc((size_t)M1TOT * KINT * 4);
    int*   ii1   = (int*)  alloc((size_t)BN * KINT * 4);
    float* dd1   = (float*)alloc((size_t)BN * KINT * 4);

    _Float16* x1loc = (_Float16*)alloc((size_t)M1TOT * 64 * 2);
    _Float16* x1    = (_Float16*)alloc((size_t)M1TOT * 128 * 2);
    _Float16* x2loc = (_Float16*)alloc((size_t)M2TOT * 256 * 2);
    _Float16* x2    = (_Float16*)alloc((size_t)M2TOT * 512 * 2);
    _Float16* xi2   = (_Float16*)alloc((size_t)M1TOT * 512 * 2);
    _Float16* xf2   = (_Float16*)alloc((size_t)M1TOT * 256 * 2);
    _Float16* xi1   = (_Float16*)alloc((size_t)BN * 256 * 2);
    _Float16* xf1   = (_Float16*)alloc((size_t)BN * 128 * 2);

    // Reused big scratch (halves):
    //  SH: H1 262144*32 | H2 65536*160 (max) | Hc2 8192*640 | Hc1 16384*288
    //  SA: T1a 262144*64 (max) | T2a 65536*128 | T3a 8192*256 | T4a 16384*128 | heads
    //  SB: T1b 262144*64 (max) | T2b 65536*256
    const size_t SH_halves = (size_t)M2TOT * KGRP * 160;
    const size_t SA_halves = (size_t)M1TOT * KGRP * 64;
    const size_t SB_halves = (size_t)M1TOT * KGRP * 64;
    _Float16* SH = (_Float16*)alloc(SH_halves * 2);
    _Float16* SA = (_Float16*)alloc(SA_halves * 2);
    _Float16* SB = (_Float16*)alloc(SB_halves * 2);

    float* out_sem = (float*)d_out;                 // [BN,8]
    float* out_emb = out_sem + (size_t)BN * 8;      // [BN,64]

    // ---- weight prep (tiny) ----
    for (int i = 0; i < 14; ++i) {
        int tot = LNT[i] * LKP[i];
        prep_w<<<cdiv_i(tot, 256), 256, 0, stream>>>(Wf[i], LK[i], LN[i], LKP[i], LNT[i], Wt[i]);
    }

    // GEMM dispatch: NT=4 strip when Nt % 64 == 0, else single-tile path.
    auto gemm = [&](const _Float16* A, int li, void* Out, int M, bool relu, bool outf32) {
        const int Kp = LKP[li], N = LN[li], Nt = LNT[li];
        if (Nt % 64 == 0) {
            dim3 g(M / 64, Nt / 64), blk(128);
            if (relu) {
                if (outf32) gemm_wmma<4, true,  true ><<<g, blk, 0, stream>>>(A, Wt[li], bf[li], Out, M, Kp, N, Nt);
                else        gemm_wmma<4, true,  false><<<g, blk, 0, stream>>>(A, Wt[li], bf[li], Out, M, Kp, N, Nt);
            } else {
                if (outf32) gemm_wmma<4, false, true ><<<g, blk, 0, stream>>>(A, Wt[li], bf[li], Out, M, Kp, N, Nt);
                else        gemm_wmma<4, false, false><<<g, blk, 0, stream>>>(A, Wt[li], bf[li], Out, M, Kp, N, Nt);
            }
        } else {
            dim3 g(M / 64, Nt / 16), blk(128);
            if (relu) {
                if (outf32) gemm_wmma<1, true,  true ><<<g, blk, 0, stream>>>(A, Wt[li], bf[li], Out, M, Kp, N, Nt);
                else        gemm_wmma<1, true,  false><<<g, blk, 0, stream>>>(A, Wt[li], bf[li], Out, M, Kp, N, Nt);
            } else {
                if (outf32) gemm_wmma<1, false, true ><<<g, blk, 0, stream>>>(A, Wt[li], bf[li], Out, M, Kp, N, Nt);
                else        gemm_wmma<1, false, false><<<g, blk, 0, stream>>>(A, Wt[li], bf[li], Out, M, Kp, N, Nt);
            }
        }
    };

    // ---- f0 = concat(x, pos) ----
    build_f0<<<cdiv_i(BN, 256), 256, 0, stream>>>(x, pos, f0h, BN);

    // ---- SA1 ----
    fps_kernel<<<BCLOUDS, 256, 0, stream>>>(pos, NPTS, M1PER, idx1);
    gather_rows_f32<<<cdiv_i(M1TOT * 3, 256), 256, 0, stream>>>(pos, idx1, M1PER, NPTS, 3, pos1, M1TOT * 3);
    knn_kernel<KGRP><<<cdiv_i(M1TOT, 128), 128, 0, stream>>>(pos, pos1, NPTS, M1PER, BCLOUDS, kidx1, kd1);
    sa_gather<<<cdiv_i(M1TOT * KGRP, 256), 256, 0, stream>>>(f0h, pos, pos1, kidx1, NPTS, M1PER, BCLOUDS, KGRP, 7, 32, SH);
    gemm(SH, 0, SA, M1TOT * KGRP, true,  false);   // relu(H1 @ W) [262144,64]
    gemm(SA, 1, SB, M1TOT * KGRP, false, false);   // [262144,64]
    maxagg<<<cdiv_i(M1TOT * 64, 256), 256, 0, stream>>>(SB, KGRP, 64, M1TOT, x1loc);
    gemm(x1loc, 2, x1, M1TOT, false, false);       // [8192,128]

    // ---- SA2 ----
    fps_kernel<<<BCLOUDS, 256, 0, stream>>>(pos1, M1PER, M2PER, idx2);
    gather_rows_f32<<<cdiv_i(M2TOT * 3, 256), 256, 0, stream>>>(pos1, idx2, M2PER, M1PER, 3, pos2, M2TOT * 3);
    knn_kernel<KGRP><<<cdiv_i(M2TOT, 128), 128, 0, stream>>>(pos1, pos2, M1PER, M2PER, BCLOUDS, kidx2, kd2);
    sa_gather<<<cdiv_i(M2TOT * KGRP, 256), 256, 0, stream>>>(x1, pos1, pos2, kidx2, M1PER, M2PER, BCLOUDS, KGRP, 128, 160, SH);
    gemm(SH, 3, SA, M2TOT * KGRP, true,  false);   // [65536,128]
    gemm(SA, 4, SB, M2TOT * KGRP, false, false);   // [65536,256]
    maxagg<<<cdiv_i(M2TOT * 256, 256), 256, 0, stream>>>(SB, KGRP, 256, M2TOT, x2loc);
    gemm(x2loc, 5, x2, M2TOT, false, false);       // [2048,512]

    // ---- FP2 ----
    knn_kernel<KINT><<<cdiv_i(M1TOT, 128), 128, 0, stream>>>(pos2, pos1, M2PER, M1PER, BCLOUDS, ii2, dd2);
    interp3<<<cdiv_i(M1TOT * 512, 256), 256, 0, stream>>>(x2, ii2, dd2, M2PER, M1PER, BCLOUDS, 512, xi2);
    concat2<<<cdiv_i(M1TOT * 640, 256), 256, 0, stream>>>(xi2, 512, x1, 128, M1TOT, 640, SH);
    gemm(SH, 6, SA, M1TOT, true,  false);
    gemm(SA, 7, xf2, M1TOT, false, false);         // [8192,256]

    // ---- FP1 ----
    knn_kernel<KINT><<<cdiv_i(BN, 128), 128, 0, stream>>>(pos1, pos, M1PER, NPTS, BCLOUDS, ii1, dd1);
    interp3<<<cdiv_i(BN * 256, 256), 256, 0, stream>>>(xf2, ii1, dd1, M1PER, NPTS, BCLOUDS, 256, xi1);
    concat2<<<cdiv_i(BN * 288, 256), 256, 0, stream>>>(xi1, 256, f0h, 7, BN, 288, SH);
    gemm(SH, 8, SA, BN, true,  false);
    gemm(SA, 9, xf1, BN, false, false);            // [16384,128]

    // ---- heads ----
    gemm(xf1, 10, SA, BN, true,  false);
    gemm(SA,  11, out_sem, BN, false, true);       // [BN,8] f32
    gemm(xf1, 12, SA, BN, true,  false);
    gemm(SA,  13, out_emb, BN, false, true);       // [BN,64] f32
}
